// SKVMN_47785806135761
// MI455X (gfx1250) — matrix operand order repaired and checked by reading, not compiled, not run on hardware
//
#include <hip/hip_runtime.h>
#include <hip/hip_bf16.h>
#include <stdint.h>

typedef __attribute__((ext_vector_type(16))) _Float16 v16h;
typedef __attribute__((ext_vector_type(8)))  float    v8f;

#define BS_   64
#define SEQ_  200
#define NROW  (BS_ * SEQ_)   // 12800
#define DS_   256
#define SM_   64
#define NSK   4096
#define KT    32
#define LDT   40             // padded LDS row stride in halves (80B rows: 16B-aligned)

// ---------------------------------------------------------------------------
// Generic WMMA GEMM:  out[m,n] = act( sum_k A[m,k] * W[n,k] + bias[n] )
// A: f16 [rows, lda] (row-major, K contiguous),  W: f16 [Dout, K] row-major.
// Block = 256 threads = 8 waves; each wave owns one 16x16 tile of a 32x64
// block tile. K loop steps by 32 through DOUBLE-BUFFERED LDS filled with
// gfx1250 async global->LDS DMA (ASYNCcnt), one tile of prefetch distance.
// Prefetch is unconditional (last one wraps to k0=0) so the steady-state
// loop is branchless: 2 async issues -> s_wait_asynccnt 2 -> barrier ->
// 4x ds_load_b128 -> v_wmma -> barrier.
// act: 0 = none, 1 = sigmoid, 2 = tanh. outF / outH optional.
// ---------------------------------------------------------------------------
__global__ __launch_bounds__(256) void wmma_gemm_kernel(
    const _Float16* __restrict__ A, int lda,
    const _Float16* __restrict__ W,
    const float* __restrict__ bias,
    float* __restrict__ outF, _Float16* __restrict__ outH,
    int K, int Dout, int act)
{
  __shared__ _Float16 sA[2][32 * LDT];
  __shared__ _Float16 sW[2][64 * LDT];

  const int tid  = threadIdx.x;
  const int wave = tid >> 5;
  const int lane = tid & 31;
  const int wr   = wave >> 2;          // 0..1 : row sub-tile
  const int wc   = wave & 3;           // 0..3 : col sub-tile
  const int rowBase = blockIdx.x * 32;
  const int colBase = blockIdx.y * 64;

  // per-thread async-copy assignments
  const int ar = tid >> 3, ac = (tid & 7) * 4;   // A tile: 32x32 halves, 8B/thread
  const int gr = tid >> 2, gc = (tid & 3) * 8;   // W tile: 64x32 halves, 16B/thread

  const _Float16* gA = A + (size_t)(rowBase + ar) * lda + ac;
  const _Float16* gW = W + (size_t)(colBase + gr) * K + gc;
  const unsigned  lA0 = (unsigned)(uintptr_t)&sA[0][ar * LDT + ac];
  const unsigned  lW0 = (unsigned)(uintptr_t)&sW[0][gr * LDT + gc];
  const unsigned  ldsStrideA = (unsigned)((uintptr_t)&sA[1][0] - (uintptr_t)&sA[0][0]);
  const unsigned  ldsStrideW = (unsigned)((uintptr_t)&sW[1][0] - (uintptr_t)&sW[0][0]);

  v8f acc = {};

  auto issue = [&](int k0, int buf) {
    unsigned la = lA0 + (unsigned)buf * ldsStrideA;
    unsigned long long ga = (unsigned long long)(uintptr_t)(gA + k0);
    asm volatile("global_load_async_to_lds_b64 %0, %1, off"
                 :: "v"(la), "v"(ga) : "memory");
    unsigned lw = lW0 + (unsigned)buf * ldsStrideW;
    unsigned long long gw = (unsigned long long)(uintptr_t)(gW + k0);
    asm volatile("global_load_async_to_lds_b128 %0, %1, off"
                 :: "v"(lw), "v"(gw) : "memory");
  };

  issue(0, 0);                         // preload tile 0
  const int nk = K / KT;

  for (int i = 0; i < nk; ++i) {
    const int buf = i & 1;
    // Unconditional prefetch of the next tile (wraps to 0 on the last
    // iteration; the dangling ops retire at S_ENDPGM's implicit wait-idle).
    const int inext = i + 1;
    issue(inext < nk ? inext * KT : 0, buf ^ 1);
    asm volatile("s_wait_asynccnt 0x2" ::: "memory");   // tile i complete
    __syncthreads();

    // Build fragments per ISA 7.12.2 (wave32, 16-bit operands).
    const int half = lane >> 4;
    const int am   = (lane & 15) + wr * 16;   // A row within tile
    const int bn   = (lane & 15) + wc * 16;   // B col within tile
    union { _Float16 h[16]; v16h v; } af, bf;
    #pragma unroll
    for (int j = 0; j < 16; j += 2) {
      int ka = half * 8 + ((j >> 3) * 16) + (j & 7);     // A: K order
      *(unsigned int*)&af.h[j] = *(const unsigned int*)&sA[buf][am * LDT + ka];
      int kb = half * 8 + ((j >> 3) * 16) + (j & 7);     // B: K order
      *(unsigned int*)&bf.h[j] = *(const unsigned int*)&sW[buf][bn * LDT + kb];
    }
    acc = __builtin_amdgcn_wmma_f32_16x16x32_f16(
        false, af.v, false, bf.v, (short)0, acc, false, false);
    __syncthreads();   // all waves done reading buf before it is refilled
  }

  const int half = lane >> 4;
  const int n = colBase + wc * 16 + (lane & 15);
  float bv = bias ? bias[n] : 0.0f;
  #pragma unroll
  for (int r = 0; r < 8; ++r) {
    int m = rowBase + wr * 16 + half * 8 + r;
    float v = acc[r] + bv;
    if (act == 1)      v = 1.0f / (1.0f + __expf(-v));
    else if (act == 2) v = tanhf(v);
    size_t o = (size_t)m * Dout + n;
    if (outF) outF[o] = v;
    if (outH) outH[o] = (_Float16)v;
  }
}

// ---------------------------------------------------------------------------
// Embedding gather + f32->f16 pack:  kv = [k | v],  rk[:,256:512] = k
// ---------------------------------------------------------------------------
__global__ __launch_bounds__(256) void gather_kernel(
    const int* __restrict__ q, const int* __restrict__ r,
    const float* __restrict__ k_emb, const float* __restrict__ x_emb,
    _Float16* __restrict__ kv, _Float16* __restrict__ rk)
{
  int n = blockIdx.x;         // b*SEQ + s
  int d = threadIdx.x;        // 0..255
  int qi = q[n];
  int vi = qi + NSK * r[n];
  float kf = k_emb[(size_t)qi * DS_ + d];
  float vf = x_emb[(size_t)vi * DS_ + d];
  kv[(size_t)n * 512 + d]        = (_Float16)kf;
  kv[(size_t)n * 512 + 256 + d]  = (_Float16)vf;
  rk[(size_t)n * 512 + 256 + d]  = (_Float16)kf;
}

__global__ void cvt_f16_kernel(const float* __restrict__ s,
                               _Float16* __restrict__ d, int n)
{
  int i = blockIdx.x * 256 + threadIdx.x;
  if (i < n) d[i] = (_Float16)s[i];
}

// ---------------------------------------------------------------------------
// Row softmax over 64 attention scores (in place); one wave per row.
// ---------------------------------------------------------------------------
__global__ __launch_bounds__(256) void softmax64_kernel(float* __restrict__ s)
{
  int row  = blockIdx.x * 8 + (threadIdx.x >> 5);
  int lane = threadIdx.x & 31;
  float* p = s + (size_t)row * SM_;
  float a = p[lane], b = p[lane + 32];
  float mx = fmaxf(a, b);
  #pragma unroll
  for (int off = 16; off > 0; off >>= 1) mx = fmaxf(mx, __shfl_xor(mx, off, 32));
  a = __expf(a - mx); b = __expf(b - mx);
  float sum = a + b;
  #pragma unroll
  for (int off = 16; off > 0; off >>= 1) sum += __shfl_xor(sum, off, 32);
  float inv = 1.0f / sum;
  p[lane] = a * inv; p[lane + 32] = b * inv;
}

// ---------------------------------------------------------------------------
// Sequential memory scan. One block per batch; thread d keeps mem[b,:,d]
// (64 floats) in VGPRs; w_t broadcast through LDS each step.
// read computed with pre-update mem; reads written as f16 into rk[:, :256].
// ---------------------------------------------------------------------------
__global__ __launch_bounds__(256) void scan_kernel(
    const float* __restrict__ w, const float* __restrict__ er,
    const float* __restrict__ ad, const float* __restrict__ Mv0,
    _Float16* __restrict__ rk)
{
  __shared__ float sw[SM_];
  int b = blockIdx.x, d = threadIdx.x;
  float m[SM_];
  #pragma unroll
  for (int j = 0; j < SM_; ++j) m[j] = Mv0[j * DS_ + d];

  for (int t = 0; t < SEQ_; ++t) {
    size_t n = (size_t)b * SEQ_ + t;
    if (d < SM_) sw[d] = w[n * SM_ + d];
    __syncthreads();
    float e = er[n * DS_ + d], a = ad[n * DS_ + d];
    float rd = 0.0f;
    #pragma unroll
    for (int j = 0; j < SM_; ++j) {
      float wj = sw[j];
      rd += wj * m[j];
      m[j] = m[j] * (1.0f - wj * e) + wj * a;
    }
    rk[n * 512 + d] = (_Float16)rd;
    __syncthreads();
  }
}

// ---------------------------------------------------------------------------
// LSTM state init (broadcast hx/cx over batch).
// ---------------------------------------------------------------------------
__global__ void lstm_init_kernel(const float* __restrict__ hx,
                                 const float* __restrict__ cx,
                                 float* __restrict__ h, _Float16* __restrict__ hh,
                                 float* __restrict__ c)
{
  int b = blockIdx.x, d = threadIdx.x;
  float hv = hx[d], cv = cx[d];
  h[b * DS_ + d] = hv; hh[b * DS_ + d] = (_Float16)hv; c[b * DS_ + d] = cv;
}

// ---------------------------------------------------------------------------
// LSTM gates + state update + output projection (LDS tree reduce for p).
// Gin already holds f@W_ih^T + b_ih; Gh holds h@W_hh^T + b_hh.
// ---------------------------------------------------------------------------
__global__ __launch_bounds__(256) void lstm_step_kernel(
    const float* __restrict__ Gin, const float* __restrict__ Gh,
    float* __restrict__ c, float* __restrict__ h, _Float16* __restrict__ hh,
    const float* __restrict__ Wp, const float* __restrict__ bp,
    float* __restrict__ pout, int t)
{
  __shared__ float red[256];
  int b = blockIdx.x, d = threadIdx.x;
  size_t n = (size_t)b * SEQ_ + t;
  const float* gi = Gin + n * 1024;
  const float* gh = Gh + (size_t)b * 1024;
  float ig = gi[d]       + gh[d];
  float fg = gi[d + 256] + gh[d + 256];
  float gg = gi[d + 512] + gh[d + 512];
  float og = gi[d + 768] + gh[d + 768];
  float si = 1.0f / (1.0f + __expf(-ig));
  float sf = 1.0f / (1.0f + __expf(-fg));
  float so = 1.0f / (1.0f + __expf(-og));
  float cv = sf * c[b * DS_ + d] + si * tanhf(gg);
  float hv = so * tanhf(cv);
  c[b * DS_ + d] = cv;
  h[b * DS_ + d] = hv;
  hh[b * DS_ + d] = (_Float16)hv;
  red[d] = hv * Wp[d];
  __syncthreads();
  #pragma unroll
  for (int off = 128; off > 0; off >>= 1) {
    if (d < off) red[d] += red[d + off];
    __syncthreads();
  }
  if (d == 0) pout[n] = 1.0f / (1.0f + __expf(-(red[0] + bp[0])));
}

// ---------------------------------------------------------------------------
extern "C" void kernel_launch(void* const* d_in, const int* in_sizes, int n_in,
                              void* d_out, int out_size, void* d_ws, size_t ws_size,
                              hipStream_t stream)
{
  const int*   q     = (const int*)  d_in[0];
  const int*   r     = (const int*)  d_in[1];
  const float* k_emb = (const float*)d_in[2];
  const float* x_emb = (const float*)d_in[3];
  const float* Mk    = (const float*)d_in[4];
  const float* Mv0   = (const float*)d_in[5];
  const float* W_a   = (const float*)d_in[6];
  const float* b_a   = (const float*)d_in[7];
  const float* W_e   = (const float*)d_in[8];
  const float* b_e   = (const float*)d_in[9];
  const float* W_add = (const float*)d_in[10];
  const float* b_add = (const float*)d_in[11];
  const float* W_f   = (const float*)d_in[12];
  const float* b_f   = (const float*)d_in[13];
  const float* hx    = (const float*)d_in[14];
  const float* cx    = (const float*)d_in[15];
  const float* W_ih  = (const float*)d_in[16];
  const float* b_ih  = (const float*)d_in[17];
  const float* W_hh  = (const float*)d_in[18];
  const float* b_hh  = (const float*)d_in[19];
  const float* W_p   = (const float*)d_in[20];
  const float* b_p   = (const float*)d_in[21];
  float* pout = (float*)d_out;

  // ---- workspace carve-up (256B aligned) ----
  char* base = (char*)d_ws;
  size_t off = 0;
  auto alloc = [&](size_t bytes) -> void* {
    void* p = base + off;
    off = (off + bytes + 255) & ~(size_t)255;
    return p;
  };
  _Float16* kv_h   = (_Float16*)alloc((size_t)NROW * 512 * 2);
  _Float16* rk_h   = (_Float16*)alloc((size_t)NROW * 512 * 2);
  _Float16* we_h   = (_Float16*)alloc((size_t)NROW * 256 * 2);
  _Float16* f_h    = (_Float16*)alloc((size_t)NROW * 256 * 2);
  _Float16* h_h    = (_Float16*)alloc((size_t)BS_  * 256 * 2);
  _Float16* Mk_h   = (_Float16*)alloc((size_t)SM_  * 256 * 2);
  _Float16* Wa_h   = (_Float16*)alloc((size_t)256  * 512 * 2);
  _Float16* We_h   = (_Float16*)alloc((size_t)256  * 256 * 2);
  _Float16* Wad_h  = (_Float16*)alloc((size_t)256  * 256 * 2);
  _Float16* Wf_h   = (_Float16*)alloc((size_t)256  * 512 * 2);
  _Float16* Wih_h  = (_Float16*)alloc((size_t)1024 * 256 * 2);
  _Float16* Whh_h  = (_Float16*)alloc((size_t)1024 * 256 * 2);
  float*    scores = (float*)   alloc((size_t)NROW * 64 * 4);
  float*    erase  = (float*)   alloc((size_t)NROW * 256 * 4);
  float*    addb   = (float*)   alloc((size_t)NROW * 256 * 4);
  float*    Gin    = (float*)   alloc((size_t)NROW * 1024 * 4);
  float*    Gh     = (float*)   alloc((size_t)BS_  * 1024 * 4);
  float*    h_f    = (float*)   alloc((size_t)BS_  * 256 * 4);
  float*    c_f    = (float*)   alloc((size_t)BS_  * 256 * 4);
  (void)ws_size; (void)in_sizes; (void)n_in; (void)out_size;

  // ---- weight conversion to f16 ----
  auto cvt = [&](const float* s, _Float16* d, int n) {
    cvt_f16_kernel<<<(n + 255) / 256, 256, 0, stream>>>(s, d, n);
  };
  cvt(Mk,    Mk_h,  SM_ * 256);
  cvt(W_a,   Wa_h,  256 * 512);
  cvt(W_e,   We_h,  256 * 256);
  cvt(W_add, Wad_h, 256 * 256);
  cvt(W_f,   Wf_h,  256 * 512);
  cvt(W_ih,  Wih_h, 1024 * 256);
  cvt(W_hh,  Whh_h, 1024 * 256);

  // ---- gathers ----
  gather_kernel<<<NROW, 256, 0, stream>>>(q, r, k_emb, x_emb, kv_h, rk_h);

  // ---- attention scores (k @ Mk^T) + softmax ----
  wmma_gemm_kernel<<<dim3(NROW / 32, SM_ / 64), 256, 0, stream>>>(
      kv_h, 512, Mk_h, nullptr, scores, nullptr, 256, SM_, 0);
  softmax64_kernel<<<NROW / 8, 256, 0, stream>>>(scores);

  // ---- we = [k|v] @ W_a^T + b_a  (needed only as f16 operand) ----
  wmma_gemm_kernel<<<dim3(NROW / 32, 256 / 64), 256, 0, stream>>>(
      kv_h, 512, Wa_h, b_a, nullptr, we_h, 512, 256, 0);

  // ---- erase = sigmoid(we @ W_e^T + b_e), add = tanh(we @ W_add^T + b_add) ----
  wmma_gemm_kernel<<<dim3(NROW / 32, 256 / 64), 256, 0, stream>>>(
      we_h, 256, We_h, b_e, erase, nullptr, 256, 256, 1);
  wmma_gemm_kernel<<<dim3(NROW / 32, 256 / 64), 256, 0, stream>>>(
      we_h, 256, Wad_h, b_add, addb, nullptr, 256, 256, 2);

  // ---- sequential memory scan -> reads (f16 into rk[:, :256]) ----
  scan_kernel<<<BS_, 256, 0, stream>>>(scores, erase, addb, Mv0, rk_h);

  // ---- f = tanh([reads|k] @ W_f^T + b_f) ----
  wmma_gemm_kernel<<<dim3(NROW / 32, 256 / 64), 256, 0, stream>>>(
      rk_h, 512, Wf_h, b_f, nullptr, f_h, 512, 256, 2);

  // ---- precompute LSTM input gates: Gin = f @ W_ih^T + b_ih ----
  wmma_gemm_kernel<<<dim3(NROW / 32, 1024 / 64), 256, 0, stream>>>(
      f_h, 256, Wih_h, b_ih, Gin, nullptr, 256, 1024, 0);

  // ---- LSTM recurrence ----
  lstm_init_kernel<<<BS_, 256, 0, stream>>>(hx, cx, h_f, h_h, c_f);
  for (int t = 0; t < SEQ_; ++t) {
    wmma_gemm_kernel<<<dim3(BS_ / 32, 1024 / 64), 256, 0, stream>>>(
        h_h, 256, Whh_h, b_hh, Gh, nullptr, 256, 1024, 0);
    lstm_step_kernel<<<BS_, 256, 0, stream>>>(
        Gin, Gh, c_f, h_f, h_h, W_p, b_p, pout, t);
  }
}
